// SparseConv2d_2267742733006
// MI455X (gfx1250) — matrix operand order repaired and checked by reading, not compiled, not run on hardware
//
#include <hip/hip_runtime.h>

typedef __attribute__((ext_vector_type(16))) __bf16 v16bf;
typedef __attribute__((ext_vector_type(8)))  float  v8f;

#define NIMG 32
#define CI   32
#define CO   32
#define HH   224
#define WW   224
#define TW   32          // output pixels (w) per block
#define TWH  (TW + 2)    // + halo
#define ROWS 16          // output rows per block
#define NTHREADS 64      // 2 waves (wave32)

__global__ __launch_bounds__(NTHREADS)
void SparseConv2d_wmma_bf16x3(const float* __restrict__ x,
                              const float* __restrict__ wgt,
                              const float* __restrict__ bias,
                              float* __restrict__ out)
{
    // LDS: input tile (hi/lo bf16), weights in WMMA A-fragment order (hi/lo), bias
    __shared__ __attribute__((aligned(32))) __bf16 xhi[3 * TWH * CI];
    __shared__ __attribute__((aligned(32))) __bf16 xlo[3 * TWH * CI];
    __shared__ __attribute__((aligned(32))) __bf16 whi[9 * 2 * 32 * 16];
    __shared__ __attribute__((aligned(32))) __bf16 wlo[9 * 2 * 32 * 16];
    __shared__ float bsh[CO];

    const int tid  = threadIdx.x;
    const int lane = tid & 31;
    const int wid  = tid >> 5;        // wave id == N-tile id (0..1)
    const int g    = lane >> 4;       // lane half-group
    const int ln   = lane & 15;

    const int gx    = blockIdx.x;
    const int wtile = gx % (WW / TW);
    const int hblk  = (gx / (WW / TW)) % (HH / ROWS);
    const int img   = gx / ((WW / TW) * (HH / ROWS));
    const int w0 = wtile * TW;
    const int h0 = hblk * ROWS;

    // ---- stage weights once: gather into A-fragment layout (contiguous k = g*16+e), split hi/lo ----
    // A fragment index j = ((r*2 + t)*32 + lane)*16 + e
    // elem e -> vgpr v=e/2, half p=e%2 ; K_local(ci) = g*16+2v+p
    for (int j = tid; j < 9 * 2 * 32 * 16; j += NTHREADS) {
        int e  = j & 15;
        int l  = (j >> 4) & 31;
        int t  = (j >> 9) & 1;
        int r  = j >> 10;
        int gg = l >> 4;
        int m  = l & 15;
        int v  = e >> 1;
        int p  = e & 1;
        int ci = gg * 16 + v * 2 + p;
        int co = t * 16 + m;
        int kh = r / 3, kw = r - kh * 3;
        float wv = wgt[((co * CI + ci) * 3 + kh) * 3 + kw];
        __bf16 hi = (__bf16)wv;
        __bf16 lo = (__bf16)(wv - (float)hi);
        whi[j] = hi;
        wlo[j] = lo;
    }
    if (tid < CO) bsh[tid] = bias[tid];

    const float* xn = x   + (size_t)img * CI * HH * WW;
    float*       on = out + (size_t)img * CO * HH * WW;

    const int pxl = wid * 16 + ln;   // pixel within block tile (B column)

    for (int hh = 0; hh < ROWS; ++hh) {
        const int h = h0 + hh;
        __syncthreads();             // all readers of LDS x done before restage

        // ---- stage 3 input rows: fp32 -> bf16 hi + residual lo; zero-fill halo ----
        for (int idx = tid; idx < 3 * CI * TWH; idx += NTHREADS) {
            int wc = idx % TWH;
            int ci = (idx / TWH) % CI;
            int r  = idx / (TWH * CI);
            int wg = w0 + wc - 1;
            int hg = h - 1 + r;
            float val = 0.0f;
            if ((unsigned)wg < WW && (unsigned)hg < HH)
                val = xn[(size_t)ci * (HH * WW) + (size_t)hg * WW + wg];
            __bf16 hi = (__bf16)val;
            __bf16 lo = (__bf16)(val - (float)hi);
            int o = (r * TWH + wc) * CI + ci;
            xhi[o] = hi;
            xlo[o] = lo;
        }
        // prefetch next iteration's new input row into cache hierarchy
        if (h + 2 < HH) {
            for (int idx = tid; idx < CI * TWH; idx += NTHREADS) {
                int wc = idx % TWH;
                int ci = idx / TWH;
                int wg = w0 + wc - 1;
                if ((unsigned)wg < WW)
                    __builtin_prefetch(&xn[(size_t)ci * (HH * WW) + (size_t)(h + 2) * WW + wg], 0, 1);
            }
        }
        __syncthreads();

        // ---- accumulators, bias preloaded (D elem (M=i+8g, N=ln)) ----
        v8f acc0, acc1;
        #pragma unroll
        for (int i = 0; i < 8; ++i) {
            acc0[i] = bsh[ 0 + g * 8 + i];
            acc1[i] = bsh[16 + g * 8 + i];
        }

        // ---- K loop: 9 taps x 32 channels; bf16x3 per tap ----
        #pragma unroll
        for (int r = 0; r < 9; ++r) {
            int kh = r / 3, kw = r - kh * 3;
            // B fragment: lane (g, ln) elem e holds K_local = g*16 + e == ci
            int bo = ((kh * TWH + pxl + kw) * CI + g * 16);
            v16bf bhi = *(const v16bf*)&xhi[bo];
            v16bf blo = *(const v16bf*)&xlo[bo];
            int a0 = ((r * 2 + 0) * 32 + lane) * 16;
            int a1 = ((r * 2 + 1) * 32 + lane) * 16;
            v16bf a0h = *(const v16bf*)&whi[a0];
            v16bf a0l = *(const v16bf*)&wlo[a0];
            v16bf a1h = *(const v16bf*)&whi[a1];
            v16bf a1l = *(const v16bf*)&wlo[a1];

            acc0 = __builtin_amdgcn_wmma_f32_16x16x32_bf16(false, a0h, false, bhi, (short)0, acc0, false, false);
            acc0 = __builtin_amdgcn_wmma_f32_16x16x32_bf16(false, a0h, false, blo, (short)0, acc0, false, false);
            acc0 = __builtin_amdgcn_wmma_f32_16x16x32_bf16(false, a0l, false, bhi, (short)0, acc0, false, false);
            acc1 = __builtin_amdgcn_wmma_f32_16x16x32_bf16(false, a1h, false, bhi, (short)0, acc1, false, false);
            acc1 = __builtin_amdgcn_wmma_f32_16x16x32_bf16(false, a1h, false, blo, (short)0, acc1, false, false);
            acc1 = __builtin_amdgcn_wmma_f32_16x16x32_bf16(false, a1l, false, bhi, (short)0, acc1, false, false);
        }

        // ---- store D tiles: co = Mtile*16 + 8g + i, pixel = w0 + pxl ----
        const int wgl = w0 + pxl;
        #pragma unroll
        for (int i = 0; i < 8; ++i) {
            int co0 = g * 8 + i;
            on[(size_t)co0 * (HH * WW) + (size_t)h * WW + wgl] = acc0[i];
            int co1 = 16 + g * 8 + i;
            on[(size_t)co1 * (HH * WW) + (size_t)h * WW + wgl] = acc1[i];
        }
    }
}

extern "C" void kernel_launch(void* const* d_in, const int* in_sizes, int n_in,
                              void* d_out, int out_size, void* d_ws, size_t ws_size,
                              hipStream_t stream) {
    const float* x    = (const float*)d_in[0];
    const float* wgt  = (const float*)d_in[1];
    const float* bias = (const float*)d_in[2];
    float* out = (float*)d_out;
    dim3 grid((WW / TW) * (HH / ROWS) * NIMG);  // 7 * 14 * 32 = 3136
    SparseConv2d_wmma_bf16x3<<<grid, NTHREADS, 0, stream>>>(x, wgt, bias, out);
}
